// Attention_84585085927925
// MI455X (gfx1250) — compile-verified, matrix operand order checked by reading, not run on hardware
//
#include <hip/hip_runtime.h>

// MI455X / gfx1250, wave32. f16 split-precision (hi+lo, ~fp32-accurate) WMMA
// GEMMs with fp32 accumulation; linear-attention rewrite (no softmax in ref):
//   out_h = SCALE * Q_h @ (K_h^T @ V_h)   -- K^T V is only 64x64.
// GEMM operands are staged global->LDS with async-to-LDS (ASYNCcnt) double
// buffering; fragments come from LDS via b128 reads; hot loop is
// ds_load_b128 + v_wmma_f32_16x16x32_f16 + global_load_async_to_lds_b128.

typedef __attribute__((ext_vector_type(16))) _Float16 v16h;
typedef __attribute__((ext_vector_type(8)))  _Float16 v8h;
typedef __attribute__((ext_vector_type(4)))  _Float16 v4h;
typedef __attribute__((ext_vector_type(8)))  float    v8f;

__device__ __forceinline__ v8f wmma_f16(v16h a, v16h b, v8f c) {
    return __builtin_amdgcn_wmma_f32_16x16x32_f16(false, a, false, b,
                                                  (short)0, c, false, false);
}

// Async copy 16B per lane from global to LDS (gfx1250, tracked by ASYNCcnt).
__device__ __forceinline__ void async_b128(unsigned lds_off, const _Float16* g) {
    asm volatile("global_load_async_to_lds_b128 %0, %1, off"
                 :: "v"(lds_off), "v"(g) : "memory");
}

template <int N>
__device__ __forceinline__ void wait_async() {
#if __has_builtin(__builtin_amdgcn_s_wait_asynccnt)
    __builtin_amdgcn_s_wait_asynccnt(N);
#else
    asm volatile("s_wait_asynccnt %0" :: "i"(N) : "memory");
#endif
}

// Fragment from LDS: 16 halves = two 16B chunks at +0 and +32 bytes
// (K-offsets {0..7} and {16..23} of a row with 80B pitch).
__device__ __forceinline__ v16h frag_lds(const char* p) {
    v8h a = *(const v8h*)p;
    v8h b = *(const v8h*)(p + 32);
    return __builtin_shufflevector(a, b, 0, 1, 2, 3, 4, 5, 6, 7,
                                   8, 9, 10, 11, 12, 13, 14, 15);
}

// Elementwise fp32 -> (hi f16, lo f16); n must be a multiple of 4*256.
__global__ __launch_bounds__(256)
void split_hl(const float* __restrict__ in, _Float16* __restrict__ h,
              _Float16* __restrict__ l) {
    const long long i = ((long long)blockIdx.x * 256 + threadIdx.x) * 4;
    v4h hh, ll;
#pragma unroll
    for (int j = 0; j < 4; ++j) {
        float x = in[i + j];
        _Float16 hv = (_Float16)x;
        hh[j] = hv;
        ll[j] = (_Float16)(x - (float)hv);
    }
    *(v4h*)(h + i) = hh;
    *(v4h*)(l + i) = ll;
}

// LDS buffer layout (bytes, per buffer; row pitch 80B = 32 halves + pad):
//   Ah: 0      .. 10240   (128 rows)
//   Al: 10240  .. 20480
//   Bh: 20480  .. 25600   (64 rows)
//   Bl: 25600  .. 30720
#define LDS_AL  10240
#define LDS_BH  20480
#define LDS_BL  25600
#define LDS_BUF 30720

// Issue the 6 async-b128 ops for this thread filling one buffer's K-tile.
__device__ __forceinline__ void async_fill(unsigned base,
                                           const _Float16* Ah, const _Float16* Al,
                                           long long lda,
                                           const _Float16* Wh, const _Float16* Wl,
                                           long long ldw,
                                           int k0, int tid) {
    // A tiles: 128 rows x 4 chunks(16B) = 512 chunks; this thread: tid, tid+256
    const int r0 = tid >> 2, c0 = (tid & 3) * 8;          // chunk -> row, k-col
    const int r1 = (tid + 256) >> 2, c1 = ((tid + 256) & 3) * 8;
    async_b128(base + r0 * 80 + c0 * 2,           Ah + (long long)r0 * lda + k0 + c0);
    async_b128(base + r1 * 80 + c1 * 2,           Ah + (long long)r1 * lda + k0 + c1);
    async_b128(base + LDS_AL + r0 * 80 + c0 * 2,  Al + (long long)r0 * lda + k0 + c0);
    async_b128(base + LDS_AL + r1 * 80 + c1 * 2,  Al + (long long)r1 * lda + k0 + c1);
    // B tiles: 64 rows x 4 chunks = 256 chunks; this thread: tid
    async_b128(base + LDS_BH + r0 * 80 + c0 * 2,  Wh + (long long)r0 * ldw + k0 + c0);
    async_b128(base + LDS_BL + r0 * 80 + c0 * 2,  Wl + (long long)r0 * ldw + k0 + c0);
}

// Generic C[m,n] = sum_k A[m,k]*W[n,k]; A,W given as f16 hi/lo pairs.
// 256 threads = 8 waves; wave owns 32x32 (2x2 WMMA tiles); block = 128x64.
// Epilogue: f32 (+bias) into Cf, or hi/lo f16 pair into Ch/Cl.
__global__ __launch_bounds__(256)
void gemm_hl(const _Float16* __restrict__ Ah, const _Float16* __restrict__ Al,
             int lda,
             const _Float16* __restrict__ Wh, const _Float16* __restrict__ Wl,
             int ldw, int K,
             float* __restrict__ Cf,
             _Float16* __restrict__ Ch, _Float16* __restrict__ Cl,
             int ldc, float scale,
             const float* __restrict__ bias,
             int nInner,
             long long aO, long long aI,
             long long wO, long long wI,
             long long cO, long long cI) {
    __shared__ __align__(16) char lds[2 * LDS_BUF];

    const int tid  = threadIdx.x;
    const int lane = tid & 31;
    const int warp = tid >> 5;
    const int wy = warp >> 1, wx = warp & 1;
    const int mBase = blockIdx.y * 128 + wy * 32;
    const int nBase = blockIdx.x * 64 + wx * 32;

    const int z  = blockIdx.z;
    const int zo = z / nInner;
    const int zi = z - zo * nInner;

    const _Float16* Abh = Ah + (long long)zo * aO + (long long)zi * aI
                             + (long long)blockIdx.y * 128 * lda;
    const _Float16* Abl = Al + (long long)zo * aO + (long long)zi * aI
                             + (long long)blockIdx.y * 128 * lda;
    const _Float16* Wbh = Wh + (long long)zo * wO + (long long)zi * wI
                             + (long long)blockIdx.x * 64 * ldw;
    const _Float16* Wbl = Wl + (long long)zo * wO + (long long)zi * wI
                             + (long long)blockIdx.x * 64 * ldw;
    const long long coff = (long long)zo * cO + (long long)zi * cI;

    const int r    = lane & 15;
    const int ksub = (lane >> 4) << 3;   // 0 or 8

    // Wave-relative LDS address = low 32 bits of the generic pointer.
    const unsigned ldsBase = (unsigned)(unsigned long long)(const void*)&lds[0];

    const int nIter = K >> 5;
    async_fill(ldsBase, Abh, Abl, lda, Wbh, Wbl, ldw, 0, tid);

    v8f acc00 = {}, acc01 = {}, acc10 = {}, acc11 = {};

    for (int it = 0; it < nIter; ++it) {
        const int cur = it & 1;
        if (it + 1 < nIter) {
            if (it > 0) __syncthreads();   // readers of buffer being refilled done
            async_fill(ldsBase + (cur ^ 1) * LDS_BUF, Abh, Abl, lda,
                       Wbh, Wbl, ldw, (it + 1) * 32, tid);
            wait_async<6>();               // oldest 6 (current buffer) retired
        } else {
            wait_async<0>();
        }
        __syncthreads();                   // all waves' fills of cur visible

        const char* buf = lds + cur * LDS_BUF;
        const char* pa0 = buf + (wy * 32 + r) * 80 + ksub * 2;
        const char* pa1 = pa0 + 16 * 80;
        const char* pb0 = buf + LDS_BH + (wx * 32 + r) * 80 + ksub * 2;
        const char* pb1 = pb0 + 16 * 80;

        v16h a0h = frag_lds(pa0);
        v16h a0l = frag_lds(pa0 + LDS_AL);
        v16h a1h = frag_lds(pa1);
        v16h a1l = frag_lds(pa1 + LDS_AL);
        v16h b0h = frag_lds(pb0);
        v16h b0l = frag_lds(pb0 + (LDS_BL - LDS_BH));
        v16h b1h = frag_lds(pb1);
        v16h b1l = frag_lds(pb1 + (LDS_BL - LDS_BH));

        // f16x3: hi*hi + hi*lo + lo*hi (dropped lo*lo ~2^-24 rel)
        acc00 = wmma_f16(a0h, b0h, acc00);
        acc00 = wmma_f16(a0h, b0l, acc00);
        acc00 = wmma_f16(a0l, b0h, acc00);

        acc01 = wmma_f16(a0h, b1h, acc01);
        acc01 = wmma_f16(a0h, b1l, acc01);
        acc01 = wmma_f16(a0l, b1h, acc01);

        acc10 = wmma_f16(a1h, b0h, acc10);
        acc10 = wmma_f16(a1h, b0l, acc10);
        acc10 = wmma_f16(a1l, b0h, acc10);

        acc11 = wmma_f16(a1h, b1h, acc11);
        acc11 = wmma_f16(a1h, b1l, acc11);
        acc11 = wmma_f16(a1l, b1h, acc11);
    }

    const int mOff = (lane >> 4) << 3;   // row offset within 16x16 C tile
    if (Cf) {
        float bv0 = 0.0f, bv1 = 0.0f;
        if (bias) {
            bv0 = bias[nBase + r];
            bv1 = bias[nBase + 16 + r];
        }
#pragma unroll
        for (int v = 0; v < 8; ++v) {
            const long long m0 = mBase + v + mOff;
            const long long m1 = m0 + 16;
            Cf[coff + m0 * ldc + nBase + r]      = scale * acc00[v] + bv0;
            Cf[coff + m0 * ldc + nBase + 16 + r] = scale * acc01[v] + bv1;
            Cf[coff + m1 * ldc + nBase + r]      = scale * acc10[v] + bv0;
            Cf[coff + m1 * ldc + nBase + 16 + r] = scale * acc11[v] + bv1;
        }
    } else {
#pragma unroll
        for (int v = 0; v < 8; ++v) {
            const long long m0 = mBase + v + mOff;
            const long long m1 = m0 + 16;
            const long long i00 = coff + m0 * ldc + nBase + r;
            const long long i01 = i00 + 16;
            const long long i10 = coff + m1 * ldc + nBase + r;
            const long long i11 = i10 + 16;
            float v00 = scale * acc00[v], v01 = scale * acc01[v];
            float v10 = scale * acc10[v], v11 = scale * acc11[v];
            _Float16 h00 = (_Float16)v00, h01 = (_Float16)v01;
            _Float16 h10 = (_Float16)v10, h11 = (_Float16)v11;
            Ch[i00] = h00; Cl[i00] = (_Float16)(v00 - (float)h00);
            Ch[i01] = h01; Cl[i01] = (_Float16)(v01 - (float)h01);
            Ch[i10] = h10; Cl[i10] = (_Float16)(v10 - (float)h10);
            Ch[i11] = h11; Cl[i11] = (_Float16)(v11 - (float)h11);
        }
    }
}

// Per (b,h): St = SCALE * (K_h^T @ V_h)^T (64x64), stored row-major ld=64 as
// hi/lo f16 so stage 3 reads K-contiguous operands. K-dim = sequence (1024):
// feature-column strided u16 gathers (small kernel, 96 blocks).
__global__ __launch_bounds__(512)
void kvT_outer(const _Float16* __restrict__ qh, const _Float16* __restrict__ ql,
               _Float16* __restrict__ Sth, _Float16* __restrict__ Stl) {
    const int bh = blockIdx.x;           // 0..95
    const int b  = bh / 12;
    const int h  = bh - b * 12;
    const int lane = threadIdx.x & 31;
    const int warp = threadIdx.x >> 5;
    const int tm = warp >> 2, tn = warp & 3;

    const int r    = lane & 15;
    const int ksub = (lane >> 4) << 3;
    const long long row = 2304;          // qkv row stride (3*D halves)

    const long long kbase = (long long)b * 1024 * row + 768  + h * 64 + tm * 16 + r;
    const long long vbase = (long long)b * 1024 * row + 1536 + h * 64 + tn * 16 + r;

    v8f acc = {};
    for (int n0 = 0; n0 < 1024; n0 += 32) {
        const _Float16* pkh = qh + kbase + (long long)(n0 + ksub) * row;
        const _Float16* pkl = ql + kbase + (long long)(n0 + ksub) * row;
        const _Float16* pvh = qh + vbase + (long long)(n0 + ksub) * row;
        const _Float16* pvl = ql + vbase + (long long)(n0 + ksub) * row;
        v16h ah, al, bhv, blv;
#pragma unroll
        for (int j = 0; j < 8; ++j) {
            ah[j]      = pkh[(long long)j * row];
            ah[j + 8]  = pkh[(long long)(j + 16) * row];
            al[j]      = pkl[(long long)j * row];
            al[j + 8]  = pkl[(long long)(j + 16) * row];
            bhv[j]     = pvh[(long long)j * row];
            bhv[j + 8] = pvh[(long long)(j + 16) * row];
            blv[j]     = pvl[(long long)j * row];
            blv[j + 8] = pvl[(long long)(j + 16) * row];
        }
        acc = wmma_f16(ah, bhv, acc);
        acc = wmma_f16(ah, blv, acc);
        acc = wmma_f16(al, bhv, acc);
    }

    // Store transposed with SCALE = sqrt(64) = 8 folded in, as hi/lo.
    const int mOff = (lane >> 4) << 3;
    const long long base = (long long)bh * 4096;
#pragma unroll
    for (int v = 0; v < 8; ++v) {
        const int m = tm * 16 + v + mOff;   // S row (q/k feature)
        const int n = tn * 16 + r;          // S col (v feature)
        const float val = 8.0f * acc[v];
        _Float16 hh = (_Float16)val;
        Sth[base + n * 64 + m] = hh;
        Stl[base + n * 64 + m] = (_Float16)(val - (float)hh);
    }
}

extern "C" void kernel_launch(void* const* d_in, const int* in_sizes, int n_in,
                              void* d_out, int out_size, void* d_ws, size_t ws_size,
                              hipStream_t stream) {
    const float* x     = (const float*)d_in[0];   // [8,1024,768]
    const float* w_qkv = (const float*)d_in[1];   // [2304,768]
    const float* w_fc  = (const float*)d_in[2];   // [768,768]
    const float* b_fc  = (const float*)d_in[3];   // [768]
    float* out = (float*)d_out;                   // [8,1024,768]

    char* ws = (char*)d_ws;
    _Float16* xh   = (_Float16*)(ws);                // 6291456 halves
    _Float16* xl   = (_Float16*)(ws + 12582912);
    _Float16* wqh  = (_Float16*)(ws + 25165824);     // 1769472 halves
    _Float16* wql  = (_Float16*)(ws + 28704768);
    _Float16* wfh  = (_Float16*)(ws + 32243712);     // 589824 halves
    _Float16* wfl  = (_Float16*)(ws + 33423360);
    _Float16* qkvh = (_Float16*)(ws + 34603008);     // 18874368 halves
    _Float16* qkvl = (_Float16*)(ws + 72351744);
    _Float16* Sth  = (_Float16*)(ws + 110100480);    // 393216 halves
    _Float16* Stl  = (_Float16*)(ws + 110886912);
    _Float16* o2h  = (_Float16*)(ws + 111673344);    // 6291456 halves
    _Float16* o2l  = (_Float16*)(ws + 124256256);

    // 0) Split fp32 inputs into f16 hi/lo (one shot).
    split_hl<<<dim3(6291456 / 1024), 256, 0, stream>>>(x, xh, xl);
    split_hl<<<dim3(1769472 / 1024), 256, 0, stream>>>(w_qkv, wqh, wql);
    split_hl<<<dim3(589824 / 1024), 256, 0, stream>>>(w_fc, wfh, wfl);

    // 1) qkv = x @ w_qkv^T   (M=8192, N=2304, K=768) -> hi/lo pair
    gemm_hl<<<dim3(2304 / 64, 8192 / 128, 1), 256, 0, stream>>>(
        xh, xl, 768, wqh, wql, 768, 768,
        nullptr, qkvh, qkvl, 2304, 1.0f, nullptr,
        1, 0, 0, 0, 0, 0, 0);

    // 2) St[bh] = SCALE * (K^T V)^T per (b,h) -> hi/lo pair
    kvT_outer<<<dim3(96), 512, 0, stream>>>(qkvh, qkvl, Sth, Stl);

    // 3) o2 = Q @ S, batched over z = b*12 + h  (M=1024, N=64, K=64)
    gemm_hl<<<dim3(1, 1024 / 128, 96), 256, 0, stream>>>(
        qkvh, qkvl, 2304, Sth, Stl, 64, 64,
        nullptr, o2h, o2l, 768, 1.0f, nullptr,
        12,
        (long long)1024 * 2304, 64,          // A: per-b, per-h offsets
        (long long)12 * 4096,   4096,        // W: St per-bh
        (long long)1024 * 768,  64);         // C: per-b, per-h offsets

    // 4) out = o2 @ w_fc^T + b_fc   (M=8192, N=768, K=768) -> f32
    gemm_hl<<<dim3(768 / 64, 8192 / 128, 1), 256, 0, stream>>>(
        o2h, o2l, 768, wfh, wfl, 768, 768,
        out, nullptr, nullptr, 768, 1.0f, b_fc,
        1, 0, 0, 0, 0, 0, 0);
}
